// DeepseekV2FlashAttention2_4252017623205
// MI455X (gfx1250) — matrix operand order repaired and checked by reading, not compiled, not run on hardware
//
#include <hip/hip_runtime.h>
#include <hip/hip_bf16.h>
#include <math.h>

// ---------- types ----------
typedef __attribute__((ext_vector_type(16))) __bf16 v16bf;
typedef __attribute__((ext_vector_type(8)))  float  v8f;
typedef __attribute__((ext_vector_type(4)))  unsigned short v4u16;

union FragBF { v16bf v; uint4 q[2]; };
union U4S    { uint4 u; unsigned short s[8]; };

static __device__ __forceinline__ unsigned short f2bf(float f) {
  unsigned int u = __float_as_uint(f);
  unsigned int r = u + 0x7FFFu + ((u >> 16) & 1u);   // round-to-nearest-even
  return (unsigned short)(r >> 16);
}
static __device__ __forceinline__ float bf2f(unsigned short u) {
  return __uint_as_float(((unsigned int)u) << 16);
}

// A-fragment (16x32 bf16): lane half lh; chunks at k = lh*8 and 16+lh*8
static __device__ __forceinline__ v16bf frag_a(const unsigned short* row, int lh) {
  FragBF f;
  f.q[0] = *(const uint4*)(row + lh * 8);
  f.q[1] = *(const uint4*)(row + 16 + lh * 8);
  return f.v;
}
// B-fragment (32x16 bf16): lane half lh; chunks at k = lh*8 and 16+lh*8
static __device__ __forceinline__ v16bf frag_b(const unsigned short* row, int lh) {
  const unsigned short* p = row + lh * 8;
  FragBF f;
  f.q[0] = *(const uint4*)(p);
  f.q[1] = *(const uint4*)(p + 16);
  return f.v;
}
static __device__ __forceinline__ v16bf ones_frag() {
  FragBF f;
  f.q[0] = make_uint4(0x3F803F80u, 0x3F803F80u, 0x3F803F80u, 0x3F803F80u);
  f.q[1] = f.q[0];
  return f.v;
}

static __device__ __forceinline__ v8f wmma_bf16(v16bf a, v16bf b, v8f c) {
  return __builtin_amdgcn_wmma_f32_16x16x32_bf16(false, a, false, b, (short)0, c,
                                                 false, false);
}

// ---------- fp32 -> bf16 bulk convert ----------
__global__ __launch_bounds__(256) void cvt_f32_bf16(const float* __restrict__ src,
                                                    unsigned short* __restrict__ dst,
                                                    int n4) {
  int id = blockIdx.x * 256 + threadIdx.x;
  if (id < n4) {
    float4 f = ((const float4*)src)[id];
    v4u16 o;
    o[0] = f2bf(f.x); o[1] = f2bf(f.y); o[2] = f2bf(f.z); o[3] = f2bf(f.w);
    ((v4u16*)dst)[id] = o;
  }
}

// ---------- generic GEMM: C[M,N] = A[M,K] @ W[N,K]^T (bf16 in, bf16 WMMA, fp32 acc) ----
// Block tile 128x64, BK=64, 256 threads = 8 waves (4m x 2n), wave tile 32x32.
// Register double-buffered global->LDS staging.
template <typename OutT>
__global__ __launch_bounds__(256) void gemm_bf16(const unsigned short* __restrict__ A,
                                                 const unsigned short* __restrict__ W,
                                                 OutT* __restrict__ C,
                                                 int M, int N, int K) {
  __shared__ alignas(16) unsigned short sA[128][72];
  __shared__ alignas(16) unsigned short sB[64][72];
  const int tid  = threadIdx.x;
  const int wave = tid >> 5;
  const int lane = tid & 31;
  const int lh   = lane >> 4;
  const int ll   = lane & 15;
  const int bm = blockIdx.y * 128;
  const int bn = blockIdx.x * 64;
  const int wm = (wave >> 1) * 32;
  const int wn = (wave & 1) * 32;

  const int ar = tid >> 3;             // A rows touched: ar, ar+32, ar+64, ar+96
  const int ac = (tid & 7) * 8;        // 16B chunk column
  const unsigned short* Abase = A + (bm + ar) * K + ac;
  const unsigned short* Wbase = W + (bn + ar) * K + ac;

  v8f acc[2][2] = {};
  uint4 ra[4], rb[2];

  // prologue: prefetch k0 = 0
#pragma unroll
  for (int i = 0; i < 4; ++i) ra[i] = *(const uint4*)(Abase + i * 32 * K);
#pragma unroll
  for (int i = 0; i < 2; ++i) rb[i] = *(const uint4*)(Wbase + i * 32 * K);

  for (int k0 = 0; k0 < K; k0 += 64) {
    __syncthreads();
#pragma unroll
    for (int i = 0; i < 4; ++i) *(uint4*)&sA[ar + 32 * i][ac] = ra[i];
#pragma unroll
    for (int i = 0; i < 2; ++i) *(uint4*)&sB[ar + 32 * i][ac] = rb[i];
    __syncthreads();

    if (k0 + 64 < K) {  // prefetch next tile while computing this one
#pragma unroll
      for (int i = 0; i < 4; ++i) ra[i] = *(const uint4*)(Abase + i * 32 * K + k0 + 64);
#pragma unroll
      for (int i = 0; i < 2; ++i) rb[i] = *(const uint4*)(Wbase + i * 32 * K + k0 + 64);
    }

#pragma unroll
    for (int kt = 0; kt < 2; ++kt) {
      v16bf af0 = frag_a(&sA[wm + ll][kt * 32], lh);
      v16bf af1 = frag_a(&sA[wm + 16 + ll][kt * 32], lh);
      v16bf bf0 = frag_b(&sB[wn + ll][kt * 32], lh);
      v16bf bf1 = frag_b(&sB[wn + 16 + ll][kt * 32], lh);
      acc[0][0] = wmma_bf16(af0, bf0, acc[0][0]);
      acc[0][1] = wmma_bf16(af0, bf1, acc[0][1]);
      acc[1][0] = wmma_bf16(af1, bf0, acc[1][0]);
      acc[1][1] = wmma_bf16(af1, bf1, acc[1][1]);
    }
  }

#pragma unroll
  for (int mi = 0; mi < 2; ++mi)
#pragma unroll
    for (int ni = 0; ni < 2; ++ni)
#pragma unroll
      for (int v = 0; v < 8; ++v) {
        int off = (bm + wm + mi * 16 + v + 8 * lh) * N + (bn + wn + ni * 16 + ll);
        if constexpr (sizeof(OutT) == 4) C[off] = acc[mi][ni][v];
        else                             C[off] = f2bf(acc[mi][ni][v]);
      }
}

// ---------- RMSNorm: one block per row, fp32 in -> bf16 out ----------
__global__ __launch_bounds__(256) void rmsnorm_bf(const float* __restrict__ src,
                                                  unsigned short* __restrict__ dst,
                                                  const float* __restrict__ w,
                                                  int D, int sstride, int dstride) {
  __shared__ float red[256];
  const int row = blockIdx.x;
  const float* s = src + row * sstride;
  unsigned short* d = dst + row * dstride;
  const int n = D >> 8;   // 2 or 6
  float x[6];
  float ss = 0.f;
  for (int i = 0; i < n; ++i) {
    x[i] = s[threadIdx.x + 256 * i];
    ss += x[i] * x[i];
  }
  red[threadIdx.x] = ss;
  __syncthreads();
  for (int off = 128; off > 0; off >>= 1) {
    if ((int)threadIdx.x < off) red[threadIdx.x] += red[threadIdx.x + off];
    __syncthreads();
  }
  float r = rsqrtf(red[0] / (float)D + 1e-6f);
  for (int i = 0; i < n; ++i) {
    int c = threadIdx.x + 256 * i;
    d[c] = f2bf(w[c] * (x[i] * r));
  }
}

// ---------- RoPE for q_pe (in-place on bf16 q, interleaved -> planar) ----------
__global__ __launch_bounds__(256) void rope_q_k(unsigned short* __restrict__ q,
                                                const int* __restrict__ pos_ids) {
  const int token = blockIdx.x;
  const float pos = (float)pos_ids[token];
  const int t = threadIdx.x;
  float x0[2], x1[2];
  int hh[2], jj[2];
#pragma unroll
  for (int i = 0; i < 2; ++i) {
    int item = t + 256 * i;            // 0..511 = 16 heads * 32 pairs
    int h = item >> 5, j = item & 31;
    hh[i] = h; jj[i] = j;
    const unsigned short* base = q + token * 3072 + h * 192 + 128;
    x0[i] = bf2f(base[2 * j]);
    x1[i] = bf2f(base[2 * j + 1]);
  }
  __syncthreads();
#pragma unroll
  for (int i = 0; i < 2; ++i) {
    int h = hh[i], j = jj[i];
    float invf = powf(10000.0f, -(float)j / 32.0f);
    float ang = pos * invf;
    float c = cosf(ang), s = sinf(ang);
    unsigned short* base = q + token * 3072 + h * 192 + 128;
    base[j]      = f2bf(x0[i] * c - x1[i] * s);
    base[32 + j] = f2bf(x1[i] * c + x0[i] * s);
  }
}

// ---------- RoPE for k_pe: ckv fp32 [...,512:576] -> kpe bf16 [token][64] planar ------
__global__ __launch_bounds__(256) void rope_k_k(const float* __restrict__ ckv,
                                                const int* __restrict__ pos_ids,
                                                unsigned short* __restrict__ kpe) {
  const int id = blockIdx.x * 256 + threadIdx.x;   // 2048*32
  const int token = id >> 5;
  const int j = id & 31;
  const float pos = (float)pos_ids[token];
  const float* s = ckv + token * 576 + 512;
  float x0 = s[2 * j], x1 = s[2 * j + 1];
  float invf = powf(10000.0f, -(float)j / 32.0f);
  float ang = pos * invf;
  float c = cosf(ang), sn = sinf(ang);
  kpe[token * 64 + j]      = f2bf(x0 * c - x1 * sn);
  kpe[token * 64 + 32 + j] = f2bf(x1 * c + x0 * sn);
}

// ---------- Flash attention (causal), block = 64 queries of one (b,h) ----------
// 128 threads = 4 waves, each wave owns 16 query rows. d=192, v=128, S=1024. bf16 io.
__global__ __launch_bounds__(128) void mla_attn(const unsigned short* __restrict__ qb,
                                                const unsigned short* __restrict__ kvb,
                                                const unsigned short* __restrict__ kpe,
                                                unsigned short* __restrict__ ob) {
  __shared__ alignas(16) unsigned short sq[64][200];
  __shared__ alignas(16) unsigned short sk[32][200];
  __shared__ alignas(16) unsigned short sv[128][40];
  __shared__ alignas(16) unsigned short sp[4][16][40];

  const int t    = threadIdx.x;
  const int wave = t >> 5;
  const int lane = t & 31;
  const int lh   = lane >> 4;
  const int ll   = lane & 15;
  const int qblk = blockIdx.x;   // 0..15
  const int h    = blockIdx.y;   // 0..15
  const int bb   = blockIdx.z;   // 0..1
  const int t0   = qblk * 64;
  const int tokbase = bb * 1024;
  const float scale = 0.07216878364870323f;  // 192^-0.5
  const v16bf vones = ones_frag();

  // stage Q tile (64 x 192): gather then scatter
  {
    uint4 qf[12];
#pragma unroll
    for (int i = 0; i < 12; ++i) {
      int id = t + 128 * i;
      int r = id / 24, c = (id % 24) * 8;
      qf[i] = *(const uint4*)(qb + (tokbase + t0 + r) * 3072 + h * 192 + c);
    }
#pragma unroll
    for (int i = 0; i < 12; ++i) {
      int id = t + 128 * i;
      int r = id / 24, c = (id % 24) * 8;
      *(uint4*)&sq[r][c] = qf[i];
    }
  }
  __syncthreads();

  v16bf fq[6];
#pragma unroll
  for (int d = 0; d < 6; ++d) fq[d] = frag_a(&sq[wave * 16 + ll][d * 32], lh);

  v8f accO[8] = {};
  float rmax[8], rsum[8];
#pragma unroll
  for (int v = 0; v < 8; ++v) { rmax[v] = -1e30f; rsum[v] = 0.f; }

  const int nkb = qblk * 2 + 2;   // key blocks of 32 covering [0, t0+63]
  for (int kb = 0; kb < nkb; ++kb) {
    // gather K (32x192) and V (32x128) chunks into registers first
    uint4 kf[6];
    U4S   vf[4];
#pragma unroll
    for (int i = 0; i < 6; ++i) {
      int id = t + 128 * i;
      int r = id / 24, c = (id % 24) * 8;
      int tok = tokbase + kb * 32 + r;
      if (c < 128) kf[i] = *(const uint4*)(kvb + tok * 4096 + h * 256 + c);
      else         kf[i] = *(const uint4*)(kpe + tok * 64 + (c - 128));
    }
#pragma unroll
    for (int i = 0; i < 4; ++i) {
      int id = t + 128 * i;
      int key = id >> 4, c = (id & 15) * 8;
      int tok = tokbase + kb * 32 + key;
      vf[i].u = *(const uint4*)(kvb + tok * 4096 + h * 256 + 128 + c);
    }
    __syncthreads();   // previous iteration's LDS readers done
#pragma unroll
    for (int i = 0; i < 6; ++i) {
      int id = t + 128 * i;
      int r = id / 24, c = (id % 24) * 8;
      *(uint4*)&sk[r][c] = kf[i];
    }
#pragma unroll
    for (int i = 0; i < 4; ++i) {
      int id = t + 128 * i;
      int key = id >> 4, c = (id & 15) * 8;
#pragma unroll
      for (int j = 0; j < 8; ++j) sv[c + j][key] = vf[i].s[j];
    }
    __syncthreads();

    // scores: 16 q rows x 32 keys, two interleaved accumulate chains
    v8f s0 = {}, s1 = {};
#pragma unroll
    for (int d = 0; d < 6; ++d) {
      v16bf bk0 = frag_b(&sk[ll][d * 32], lh);
      v16bf bk1 = frag_b(&sk[16 + ll][d * 32], lh);
      s0 = wmma_bf16(fq[d], bk0, s0);
      s1 = wmma_bf16(fq[d], bk1, s1);
    }
    v8f sc[2]; sc[0] = s0; sc[1] = s1;

    // online softmax (per row = VGPR v + half)
    float nmax[8];
#pragma unroll
    for (int v = 0; v < 8; ++v) {
      int qpos = t0 + wave * 16 + v + 8 * lh;
      float m = rmax[v];
#pragma unroll
      for (int nt = 0; nt < 2; ++nt) {
        int kpos = kb * 32 + nt * 16 + ll;
        float x = sc[nt][v] * scale;
        x = (kpos <= qpos) ? x : -1e30f;
        sc[nt][v] = x;
        m = fmaxf(m, x);
      }
#pragma unroll
      for (int off = 8; off > 0; off >>= 1) m = fmaxf(m, __shfl_xor(m, off, 32));
      nmax[v] = m;
    }
    float alpha[8];
#pragma unroll
    for (int v = 0; v < 8; ++v) {
      float m = nmax[v];
      alpha[v] = __expf(rmax[v] - m);
      rmax[v] = m;
#pragma unroll
      for (int nt = 0; nt < 2; ++nt) {
        float p = __expf(sc[nt][v] - m);
        sp[wave][v + 8 * lh][nt * 16 + ll] = f2bf(p);
      }
#pragma unroll
      for (int vt = 0; vt < 8; ++vt) accO[vt][v] *= alpha[v];
    }
    __syncthreads();   // P LDS write -> fragment read ordering

    // O += P @ V ; row-sum of P via P @ ones (free softmax denominator)
    v16bf pa = frag_a(&sp[wave][ll][0], lh);
    v8f zr = {};
    v8f rs = wmma_bf16(pa, vones, zr);   // rs[v] = sum_k P[row v+8lh][k]
#pragma unroll
    for (int v = 0; v < 8; ++v) rsum[v] = rsum[v] * alpha[v] + rs[v];
#pragma unroll
    for (int vt = 0; vt < 8; ++vt) {
      v16bf vb = frag_b(&sv[vt * 16 + ll][0], lh);
      accO[vt] = wmma_bf16(pa, vb, accO[vt]);
    }
  }

  // write output [token][h*128 + vdim] (bf16)
#pragma unroll
  for (int vt = 0; vt < 8; ++vt)
#pragma unroll
    for (int v = 0; v < 8; ++v) {
      int m = v + 8 * lh;
      int token = tokbase + t0 + wave * 16 + m;
      ob[token * 2048 + h * 128 + vt * 16 + ll] = f2bf(accO[vt][v] / rsum[v]);
    }
}

// ---------- host-side launcher ----------
extern "C" void kernel_launch(void* const* d_in, const int* in_sizes, int n_in,
                              void* d_out, int out_size, void* d_ws, size_t ws_size,
                              hipStream_t stream) {
  (void)in_sizes; (void)n_in; (void)out_size; (void)ws_size;
  const float* hidden  = (const float*)d_in[0];  // [2,1024,2048]
  const float* q_a_w   = (const float*)d_in[1];  // [1536,2048]
  const float* q_a_ln  = (const float*)d_in[2];  // [1536]
  const float* q_b_w   = (const float*)d_in[3];  // [3072,1536]
  const float* kv_a_w  = (const float*)d_in[4];  // [576,2048]
  const float* kv_a_ln = (const float*)d_in[5];  // [512]
  const float* kv_b_w  = (const float*)d_in[6];  // [4096,512]
  const float* o_w     = (const float*)d_in[7];  // [2048,2048]
  const int*   pos     = (const int*)d_in[8];    // [2,1024]
  float* out = (float*)d_out;

  typedef unsigned short u16;
  char* ws = (char*)d_ws;
  float* q_lat   = (float*)ws; ws += (size_t)2048 * 1536 * 4;
  float* ckv     = (float*)ws; ws += (size_t)2048 * 576 * 4;
  u16* hidden_bf = (u16*)ws;   ws += (size_t)2048 * 2048 * 2;
  u16* qaw_bf    = (u16*)ws;   ws += (size_t)1536 * 2048 * 2;
  u16* qbw_bf    = (u16*)ws;   ws += (size_t)3072 * 1536 * 2;
  u16* kvaw_bf   = (u16*)ws;   ws += (size_t)576 * 2048 * 2;
  u16* kvbw_bf   = (u16*)ws;   ws += (size_t)4096 * 512 * 2;
  u16* ow_bf     = (u16*)ws;   ws += (size_t)2048 * 2048 * 2;
  u16* qlat_bf   = (u16*)ws;   ws += (size_t)2048 * 1536 * 2;
  u16* ckvn_bf   = (u16*)ws;   ws += (size_t)2048 * 512 * 2;
  u16* kpe_bf    = (u16*)ws;   ws += (size_t)2048 * 64 * 2;
  u16* q_bf      = (u16*)ws;   ws += (size_t)2048 * 3072 * 2;
  u16* kv_bf     = (u16*)ws;   ws += (size_t)2048 * 4096 * 2;
  u16* attn_bf   = (u16*)ws;   ws += (size_t)2048 * 2048 * 2;

  // one-time bf16 conversions (inputs + weights)
  cvt_f32_bf16<<<4096, 256, 0, stream>>>(hidden, hidden_bf, 2048 * 2048 / 4);
  cvt_f32_bf16<<<3072, 256, 0, stream>>>(q_a_w,  qaw_bf,  1536 * 2048 / 4);
  cvt_f32_bf16<<<4608, 256, 0, stream>>>(q_b_w,  qbw_bf,  3072 * 1536 / 4);
  cvt_f32_bf16<<<1152, 256, 0, stream>>>(kv_a_w, kvaw_bf, 576 * 2048 / 4);
  cvt_f32_bf16<<<2048, 256, 0, stream>>>(kv_b_w, kvbw_bf, 4096 * 512 / 4);
  cvt_f32_bf16<<<4096, 256, 0, stream>>>(o_w,    ow_bf,   2048 * 2048 / 4);

  // latent projections (fp32 out for exact-norm path)
  gemm_bf16<float><<<dim3(1536 / 64, 2048 / 128), 256, 0, stream>>>(
      hidden_bf, qaw_bf, q_lat, 2048, 1536, 2048);
  gemm_bf16<float><<<dim3(576 / 64, 2048 / 128), 256, 0, stream>>>(
      hidden_bf, kvaw_bf, ckv, 2048, 576, 2048);
  // norms (bf16 out) + k_pe rope
  rmsnorm_bf<<<2048, 256, 0, stream>>>(q_lat, qlat_bf, q_a_ln, 1536, 1536, 1536);
  rmsnorm_bf<<<2048, 256, 0, stream>>>(ckv, ckvn_bf, kv_a_ln, 512, 576, 512);
  rope_k_k<<<(2048 * 32) / 256, 256, 0, stream>>>(ckv, pos, kpe_bf);
  // up-projections (bf16 out) + q rope
  gemm_bf16<u16><<<dim3(3072 / 64, 2048 / 128), 256, 0, stream>>>(
      qlat_bf, qbw_bf, q_bf, 2048, 3072, 1536);
  rope_q_k<<<2048, 256, 0, stream>>>(q_bf, pos);
  gemm_bf16<u16><<<dim3(4096 / 64, 2048 / 128), 256, 0, stream>>>(
      ckvn_bf, kvbw_bf, kv_bf, 2048, 4096, 512);
  // attention (bf16 out)
  mla_attn<<<dim3(16, 16, 2), 128, 0, stream>>>(q_bf, kv_bf, kpe_bf, attn_bf);
  // output projection (fp32 out)
  gemm_bf16<float><<<dim3(2048 / 64, 2048 / 128), 256, 0, stream>>>(
      attn_bf, ow_bf, out, 2048, 2048, 2048);
}
